// GRUEncoder_42623255445795
// MI455X (gfx1250) — compile-verified, matrix operand order checked by reading, not run on hardware
//
#include <hip/hip_runtime.h>

// ---------------------------------------------------------------------------
// 2-layer GRU encoder, fully fused per 32-row batch tile.
//   B=256, T=512, IN=75 (padded to 96), H=256, gates G=768.
// bf16 WMMA (v_wmma_f32_16x16x32_bf16) with f32 accumulation.
// - Weight fragments re-streamed from L2 each timestep through laundered
//   SGPR pointers (blocks LICM hoist+spill) with explicit addrspace(1)
//   loads (global_load_b128, no flat/DScnt coupling).
// - Addresses computed inline from the per-iteration laundered base: keeps
//   fragment live ranges short (hoisting offsets caused spills -- reverted).
// - M-tile = 32 rows: each B fragment feeds 2 WMMAs (halves weight traffic).
// ---------------------------------------------------------------------------

typedef __attribute__((ext_vector_type(16))) __bf16 v16bf;
typedef __attribute__((ext_vector_type(8)))  float  v8f;

#define B_    256
#define T_    512
#define H_    256
#define G_    768            // 3*H
#define KIN_  96             // 75 padded up to 3*32
#define BT_   (B_ * T_)
#define MT_   32             // batch rows per workgroup (2 x 16 M-subtiles)

// Opaque SGPR launder: loads through the result cannot be hoisted out of the
// t-loop (prevents LICM + scratch spilling of ~80 weight fragments).
#define LAUNDER_U(u) asm volatile("" : "+s"(u))

// Global-address-space vector load (emits global_load_b128 pair).
typedef const __attribute__((address_space(1))) v16bf* gv16p;
__device__ __forceinline__ v16bf ldg16(uintptr_t byte_addr) {
  return *(gv16p)byte_addr;
}

// ------------------------- bf16 convert / pad ------------------------------
__global__ void cvt_pad_bf16(const float* __restrict__ src, __bf16* __restrict__ dst,
                             int rows, int K, int Kp) {
  int idx = blockIdx.x * blockDim.x + threadIdx.x;
  if (idx >= rows * Kp) return;
  int r = idx / Kp;
  int k = idx - r * Kp;
  float v = (k < K) ? src[(size_t)r * K + k] : 0.0f;
  dst[idx] = (__bf16)v;
}

__device__ __forceinline__ float sigmoid_f(float v) {
  return 1.0f / (1.0f + __expf(-v));
}
__device__ __forceinline__ float tanh_f(float v) {
  return 2.0f / (1.0f + __expf(-2.0f * v)) - 1.0f;
}

#define WMMA_BF16(A, Bm, C) \
  __builtin_amdgcn_wmma_f32_16x16x32_bf16(false, (A), false, (Bm), (short)0, (C), false, false)

// ------------------------- fused persistent GRU ----------------------------
// grid = B_/32 workgroups, 512 threads (16 waves). Wave w owns hidden cols
// [16w, 16w+16) for both 16-row M-subtiles; computes all three gates.
// h0/h1 double-buffered in LDS as bf16.
__launch_bounds__(512, 1)
__global__ void gru2_persistent(
    const __bf16* __restrict__ xbf,    // [B*T, KIN_]
    const __bf16* __restrict__ wih0,   // [G_, KIN_]
    const __bf16* __restrict__ whh0,   // [G_, H_]
    const __bf16* __restrict__ wih1,   // [G_, H_]
    const __bf16* __restrict__ whh1,   // [G_, H_]
    const __bf16* __restrict__ fcw,    // [H_, H_]
    const float* __restrict__ b_ih0, const float* __restrict__ b_hh0,
    const float* __restrict__ b_ih1, const float* __restrict__ b_hh1,
    const float* __restrict__ fc_b,
    float* __restrict__ out)           // [B_, H_]
{
  __shared__ __bf16 h0s[2][MT_ * H_];  // 16 KB per buffer
  __shared__ __bf16 h1s[2][MT_ * H_];

  const int tid  = threadIdx.x;
  const int wave = tid >> 5;           // 0..15 (wave32)
  const int lane = tid & 31;
  const int row  = lane & 15;          // A-row / B-col / C-col index
  const int half = lane >> 4;          // K sub-chunk selector
  const int n0   = wave << 4;          // hidden column tile base
  const int b0   = blockIdx.x * MT_;   // batch tile base

  for (int i = tid; i < MT_ * H_; i += 512) {
    h0s[0][i] = (__bf16)0.0f; h0s[1][i] = (__bf16)0.0f;
    h1s[0][i] = (__bf16)0.0f; h1s[1][i] = (__bf16)0.0f;
  }
  __syncthreads();

  // Hoist per-column biases (loop-invariant over t, tiny).
  float bi0[3], bh0[3], bi1[3], bh1[3];
#pragma unroll
  for (int g = 0; g < 3; g++) {
    bi0[g] = b_ih0[g * H_ + n0 + row];
    bh0[g] = b_hh0[g * H_ + n0 + row];
    bi1[g] = b_ih1[g * H_ + n0 + row];
    bh1[g] = b_hh1[g * H_ + n0 + row];
  }

  const size_t xrow0 = (size_t)(b0 + row)      * T_;   // M-subtile 0
  const size_t xrow1 = (size_t)(b0 + 16 + row) * T_;   // M-subtile 1
  const v8f vzero = {};
  int p = 0;

  for (int t = 0; t < T_; t++) {
    // Opaque per-iteration weight bases: re-stream from (L2-hot) global.
    uintptr_t uwih0 = (uintptr_t)wih0; LAUNDER_U(uwih0);
    uintptr_t uwhh0 = (uintptr_t)whh0; LAUNDER_U(uwhh0);
    uintptr_t uwih1 = (uintptr_t)wih1; LAUNDER_U(uwih1);
    uintptr_t uwhh1 = (uintptr_t)whh1; LAUNDER_U(uwhh1);

    if (t + 1 < T_) {  // pull next timestep's x tiles toward L2 (global_prefetch_b8)
      __builtin_prefetch(&xbf[(xrow0 + t + 1) * KIN_], 0, 3);
      __builtin_prefetch(&xbf[(xrow1 + t + 1) * KIN_], 0, 3);
    }

    // ======================= layer 0 =======================
    v8f ax[2][3], ah[2][3];
#pragma unroll
    for (int mi = 0; mi < 2; mi++)
#pragma unroll
      for (int g = 0; g < 3; g++) { ax[mi][g] = vzero; ah[mi][g] = vzero; }

    // xg0 = x_t @ W_ih0^T  (K = 96)
#pragma unroll
    for (int kk = 0; kk < 3; kk++) {
      v16bf a0 = *(const v16bf*)&xbf[(xrow0 + t) * KIN_ + kk * 32 + half * 16];
      v16bf a1 = *(const v16bf*)&xbf[(xrow1 + t) * KIN_ + kk * 32 + half * 16];
#pragma unroll
      for (int g = 0; g < 3; g++) {
        v16bf b = ldg16(uwih0 + 2u * ((size_t)(g * H_ + n0 + row) * KIN_ + kk * 32 + half * 16));
        ax[0][g] = WMMA_BF16(a0, b, ax[0][g]);
        ax[1][g] = WMMA_BF16(a1, b, ax[1][g]);
      }
    }
    // hg0 = h0 @ W_hh0^T  (K = 256)
#pragma unroll
    for (int kk = 0; kk < 8; kk++) {
      v16bf a0 = *(const v16bf*)&h0s[p][(row)      * H_ + kk * 32 + half * 16];
      v16bf a1 = *(const v16bf*)&h0s[p][(16 + row) * H_ + kk * 32 + half * 16];
#pragma unroll
      for (int g = 0; g < 3; g++) {
        v16bf b = ldg16(uwhh0 + 2u * ((size_t)(g * H_ + n0 + row) * H_ + kk * 32 + half * 16));
        ah[0][g] = WMMA_BF16(a0, b, ah[0][g]);
        ah[1][g] = WMMA_BF16(a1, b, ah[1][g]);
      }
    }
    // gate math; write h0_new into the other LDS buffer
#pragma unroll
    for (int mi = 0; mi < 2; mi++) {
#pragma unroll
      for (int i = 0; i < 8; i++) {
        int m = mi * 16 + i + 8 * half;
        float r = sigmoid_f(ax[mi][0][i] + bi0[0] + ah[mi][0][i] + bh0[0]);
        float z = sigmoid_f(ax[mi][1][i] + bi0[1] + ah[mi][1][i] + bh0[1]);
        float n = tanh_f(ax[mi][2][i] + bi0[2] + r * (ah[mi][2][i] + bh0[2]));
        float hp = (float)h0s[p][m * H_ + n0 + row];
        h0s[p ^ 1][m * H_ + n0 + row] = (__bf16)((1.0f - z) * n + z * hp);
      }
    }
    __syncthreads();

    // ======================= layer 1 =======================
    v8f cx[2][3], ch[2][3];
#pragma unroll
    for (int mi = 0; mi < 2; mi++)
#pragma unroll
      for (int g = 0; g < 3; g++) { cx[mi][g] = vzero; ch[mi][g] = vzero; }

    // x-path: A = freshly produced h0_new
#pragma unroll
    for (int kk = 0; kk < 8; kk++) {
      v16bf a0 = *(const v16bf*)&h0s[p ^ 1][(row)      * H_ + kk * 32 + half * 16];
      v16bf a1 = *(const v16bf*)&h0s[p ^ 1][(16 + row) * H_ + kk * 32 + half * 16];
#pragma unroll
      for (int g = 0; g < 3; g++) {
        v16bf b = ldg16(uwih1 + 2u * ((size_t)(g * H_ + n0 + row) * H_ + kk * 32 + half * 16));
        cx[0][g] = WMMA_BF16(a0, b, cx[0][g]);
        cx[1][g] = WMMA_BF16(a1, b, cx[1][g]);
      }
    }
    // h-path: A = previous h1
#pragma unroll
    for (int kk = 0; kk < 8; kk++) {
      v16bf a0 = *(const v16bf*)&h1s[p][(row)      * H_ + kk * 32 + half * 16];
      v16bf a1 = *(const v16bf*)&h1s[p][(16 + row) * H_ + kk * 32 + half * 16];
#pragma unroll
      for (int g = 0; g < 3; g++) {
        v16bf b = ldg16(uwhh1 + 2u * ((size_t)(g * H_ + n0 + row) * H_ + kk * 32 + half * 16));
        ch[0][g] = WMMA_BF16(a0, b, ch[0][g]);
        ch[1][g] = WMMA_BF16(a1, b, ch[1][g]);
      }
    }
#pragma unroll
    for (int mi = 0; mi < 2; mi++) {
#pragma unroll
      for (int i = 0; i < 8; i++) {
        int m = mi * 16 + i + 8 * half;
        float r = sigmoid_f(cx[mi][0][i] + bi1[0] + ch[mi][0][i] + bh1[0]);
        float z = sigmoid_f(cx[mi][1][i] + bi1[1] + ch[mi][1][i] + bh1[1]);
        float n = tanh_f(cx[mi][2][i] + bi1[2] + r * (ch[mi][2][i] + bh1[2]));
        float hp = (float)h1s[p][m * H_ + n0 + row];
        h1s[p ^ 1][m * H_ + n0 + row] = (__bf16)((1.0f - z) * n + z * hp);
      }
    }
    __syncthreads();
    p ^= 1;
  }

  // ======================= final FC: emb = h1_T @ fc_W^T + fc_b ===========
  v8f acc[2] = {vzero, vzero};
#pragma unroll
  for (int kk = 0; kk < 8; kk++) {
    v16bf a0 = *(const v16bf*)&h1s[p][(row)      * H_ + kk * 32 + half * 16];
    v16bf a1 = *(const v16bf*)&h1s[p][(16 + row) * H_ + kk * 32 + half * 16];
    v16bf b  = *(const v16bf*)&fcw[(size_t)(n0 + row) * H_ + kk * 32 + half * 16];
    acc[0] = WMMA_BF16(a0, b, acc[0]);
    acc[1] = WMMA_BF16(a1, b, acc[1]);
  }
  float fb = fc_b[n0 + row];
#pragma unroll
  for (int mi = 0; mi < 2; mi++) {
#pragma unroll
    for (int i = 0; i < 8; i++) {
      int m = mi * 16 + i + 8 * half;
      out[(size_t)(b0 + m) * H_ + n0 + row] = acc[mi][i] + fb;
    }
  }
}

// ---------------------------------------------------------------------------
extern "C" void kernel_launch(void* const* d_in, const int* in_sizes, int n_in,
                              void* d_out, int out_size, void* d_ws, size_t ws_size,
                              hipStream_t stream) {
  const float* x     = (const float*)d_in[0];   // [B,T,25,3] == [B*T,75]
  const float* W_ih0 = (const float*)d_in[1];   // [768,75]
  const float* W_hh0 = (const float*)d_in[2];   // [768,256]
  const float* b_ih0 = (const float*)d_in[3];
  const float* b_hh0 = (const float*)d_in[4];
  const float* W_ih1 = (const float*)d_in[5];   // [768,256]
  const float* W_hh1 = (const float*)d_in[6];   // [768,256]
  const float* b_ih1 = (const float*)d_in[7];
  const float* b_hh1 = (const float*)d_in[8];
  const float* fc_W  = (const float*)d_in[9];   // [256,256]
  const float* fc_b  = (const float*)d_in[10];
  float* out = (float*)d_out;

  char* ws = (char*)d_ws;
  size_t o = 0;
  __bf16* xbf  = (__bf16*)(ws + o); o += (size_t)BT_ * KIN_ * 2;  // 25.2 MB
  __bf16* wih0 = (__bf16*)(ws + o); o += (size_t)G_ * KIN_ * 2;
  __bf16* whh0 = (__bf16*)(ws + o); o += (size_t)G_ * H_ * 2;
  __bf16* wih1 = (__bf16*)(ws + o); o += (size_t)G_ * H_ * 2;
  __bf16* whh1 = (__bf16*)(ws + o); o += (size_t)G_ * H_ * 2;
  __bf16* fcw  = (__bf16*)(ws + o); o += (size_t)H_ * H_ * 2;

  const int thr = 256;
  // x: [B*T,75] -> bf16 padded [B*T,96]
  cvt_pad_bf16<<<(BT_ * KIN_ + thr - 1) / thr, thr, 0, stream>>>(x, xbf, BT_, 75, KIN_);
  cvt_pad_bf16<<<(G_ * KIN_ + thr - 1) / thr, thr, 0, stream>>>(W_ih0, wih0, G_, 75, KIN_);
  cvt_pad_bf16<<<(G_ * H_ + thr - 1) / thr, thr, 0, stream>>>(W_hh0, whh0, G_, H_, H_);
  cvt_pad_bf16<<<(G_ * H_ + thr - 1) / thr, thr, 0, stream>>>(W_ih1, wih1, G_, H_, H_);
  cvt_pad_bf16<<<(G_ * H_ + thr - 1) / thr, thr, 0, stream>>>(W_hh1, whh1, G_, H_, H_);
  cvt_pad_bf16<<<(H_ * H_ + thr - 1) / thr, thr, 0, stream>>>(fc_W, fcw, H_, H_, H_);

  gru2_persistent<<<B_ / MT_, 512, 0, stream>>>(
      xbf, wih0, whh0, wih1, whh1, fcw,
      b_ih0, b_hh0, b_ih1, b_hh1, fc_b, out);
}
